// ComplexAttention_12060268167990
// MI455X (gfx1250) — compile-verified, matrix operand order checked by reading, not run on hardware
//
#include <hip/hip_runtime.h>

// ---------------------------------------------------------------------------
// Problem constants (from reference): B=2, T=2048, DM=1024, H=16, D=64
// ---------------------------------------------------------------------------
#define BB   2
#define TT   2048
#define DM   1024
#define HH   16
#define DD   64
#define HD   (HH * DD)          // 1024
#define MROWS (BB * TT)         // 4096
#define SCALE 0.125f            // D^-0.5
#define LOG2E 1.44269504088896f

typedef __attribute__((ext_vector_type(16))) __bf16 v16bf;
typedef __attribute__((ext_vector_type(8)))  float  v8f;

union FragA { v16bf v; uint4 q[2]; };
union U8    { uint4 u; __bf16 b[8]; };

static __device__ inline uint4 ldg16(const __bf16* p) {
    return *reinterpret_cast<const uint4*>(p);
}
static __device__ inline uint4 lds16(const __bf16* p) {
    return *reinterpret_cast<const uint4*>(p);
}
static __device__ inline v8f wmma_bf16(const FragA& a, const FragA& b, v8f c) {
    return __builtin_amdgcn_wmma_f32_16x16x32_bf16(false, a.v, false, b.v,
                                                   (short)0, c, false, false);
}
static __device__ inline v8f vzero8() {
    v8f z;
    #pragma unroll
    for (int i = 0; i < 8; i++) z[i] = 0.0f;
    return z;
}

// ---------------------------------------------------------------------------
// CDNA5 async global->LDS copy (16B per lane), tracked by ASYNCcnt.
// LDS offset = low 32 bits of the generic shared-aperture address.
// ---------------------------------------------------------------------------
static __device__ inline void async_cp16(const __bf16* g, __bf16* l) {
    unsigned       lo = (unsigned)(unsigned long long)l;
    unsigned long long ga = (unsigned long long)g;
    asm volatile("global_load_async_to_lds_b128 %0, %1, off"
                 :: "v"(lo), "v"(ga) : "memory");
}
static __device__ inline void wait_async0() {
#if __has_builtin(__builtin_amdgcn_s_wait_asynccnt)
    __builtin_amdgcn_s_wait_asynccnt(0);
#else
    asm volatile("s_wait_asynccnt 0" ::: "memory");
#endif
}

// ---------------------------------------------------------------------------
// f32 -> bf16 conversion (x)
// ---------------------------------------------------------------------------
__global__ __launch_bounds__(256)
void cvt_bf16_kernel(const float* __restrict__ src, __bf16* __restrict__ dst, int n) {
    int i = blockIdx.x * 256 + threadIdx.x;
    if (i < n) dst[i] = (__bf16)src[i];
}

// ---------------------------------------------------------------------------
// f32 [1024][1024] -> transposed bf16 [n][k] via 32x32 LDS tile (coalesced
// both sides). Lets GEMM B-tiles be staged with pure async 16B copies.
// ---------------------------------------------------------------------------
__global__ __launch_bounds__(256)
void cvt_t_kernel(const float* __restrict__ src, __bf16* __restrict__ dst) {
    __shared__ float tile[32][33];
    const int k0 = blockIdx.x * 32, n0 = blockIdx.y * 32;
    const int tx = threadIdx.x & 31, ty = threadIdx.x >> 5;  // ty: 0..7
    #pragma unroll
    for (int i = 0; i < 4; i++) {
        int k = ty * 4 + i;
        tile[k][tx] = src[(k0 + k) * 1024 + n0 + tx];
    }
    __syncthreads();
    #pragma unroll
    for (int i = 0; i < 4; i++) {
        int n = ty * 4 + i;
        dst[(n0 + n) * 1024 + k0 + tx] = (__bf16)tile[tx][n];
    }
}

// ---------------------------------------------------------------------------
// Paired projection GEMM + RoPE epilogue (async double-buffered LDS).
//   out_{re,im}[b,h,t,d] = rope(x @ Wre, x @ Wim) * outScale      (bf16)
// Weights arrive pre-transposed [n][k]. Block: 128 thr, tile 64x64, K-step 32.
// ---------------------------------------------------------------------------
__global__ __launch_bounds__(128)
void proj_pair_rope_kernel(const __bf16* __restrict__ xb,
                           const __bf16* __restrict__ WreT,
                           const __bf16* __restrict__ WimT,
                           const float*  __restrict__ cosp,
                           const float*  __restrict__ sinp,
                           __bf16* __restrict__ outRe,
                           __bf16* __restrict__ outIm,
                           float outScale)
{
    __shared__ __align__(16) __bf16 AsB [2][64 * 32];   // [m][k]
    __shared__ __align__(16) __bf16 BreB[2][64 * 32];   // [n][k]
    __shared__ __align__(16) __bf16 BimB[2][64 * 32];

    const int tid  = threadIdx.x;
    const int lane = tid & 31, w = tid >> 5;
    const int half = lane >> 4, l16 = lane & 15;
    const int m0 = blockIdx.x * 64;
    const int n0 = blockIdx.y * 64;

    const int c0 = tid, c1 = tid + 128;          // chunk ids (256 total)
    const int ar0 = c0 >> 2, ak0 = (c0 & 3) << 3;
    const int ar1 = c1 >> 2, ak1 = (c1 & 3) << 3;

    v8f accRe[4], accIm[4];
    #pragma unroll
    for (int i = 0; i < 4; i++) { accRe[i] = vzero8(); accIm[i] = vzero8(); }

    // issue first stage
    {
        async_cp16(&xb[(m0 + ar0) * DM + ak0],        &AsB [0][ar0 * 32 + ak0]);
        async_cp16(&xb[(m0 + ar1) * DM + ak1],        &AsB [0][ar1 * 32 + ak1]);
        async_cp16(&WreT[(n0 + ar0) * DM + ak0],      &BreB[0][ar0 * 32 + ak0]);
        async_cp16(&WreT[(n0 + ar1) * DM + ak1],      &BreB[0][ar1 * 32 + ak1]);
        async_cp16(&WimT[(n0 + ar0) * DM + ak0],      &BimB[0][ar0 * 32 + ak0]);
        async_cp16(&WimT[(n0 + ar1) * DM + ak1],      &BimB[0][ar1 * 32 + ak1]);
    }

    for (int kb = 0; kb < DM; kb += 32) {
        const int buf = (kb >> 5) & 1;
        wait_async0();
        __syncthreads();
        if (kb + 32 < DM) {
            const int nb = buf ^ 1, kn = kb + 32;
            async_cp16(&xb[(m0 + ar0) * DM + kn + ak0],   &AsB [nb][ar0 * 32 + ak0]);
            async_cp16(&xb[(m0 + ar1) * DM + kn + ak1],   &AsB [nb][ar1 * 32 + ak1]);
            async_cp16(&WreT[(n0 + ar0) * DM + kn + ak0], &BreB[nb][ar0 * 32 + ak0]);
            async_cp16(&WreT[(n0 + ar1) * DM + kn + ak1], &BreB[nb][ar1 * 32 + ak1]);
            async_cp16(&WimT[(n0 + ar0) * DM + kn + ak0], &BimB[nb][ar0 * 32 + ak0]);
            async_cp16(&WimT[(n0 + ar1) * DM + kn + ak1], &BimB[nb][ar1 * 32 + ak1]);
        }

        const __bf16* A  = AsB [buf];
        const __bf16* Br = BreB[buf];
        const __bf16* Bi = BimB[buf];

        FragA a;
        a.q[0] = lds16(&A[(w * 16 + l16) * 32 + half * 8]);
        a.q[1] = lds16(&A[(w * 16 + l16) * 32 + half * 8 + 16]);
        FragA fr[4], fi[4];
        #pragma unroll
        for (int nf = 0; nf < 4; nf++) {
            const int nn = nf * 16 + l16;
            fr[nf].q[0] = lds16(&Br[nn * 32 + half * 8]);
            fr[nf].q[1] = lds16(&Br[nn * 32 + half * 8 + 16]);
            fi[nf].q[0] = lds16(&Bi[nn * 32 + half * 8]);
            fi[nf].q[1] = lds16(&Bi[nn * 32 + half * 8 + 16]);
        }
        #pragma unroll
        for (int nf = 0; nf < 4; nf++) {
            accRe[nf] = wmma_bf16(a, fr[nf], accRe[nf]);
            accIm[nf] = wmma_bf16(a, fi[nf], accIm[nf]);
        }
    }

    // ---- RoPE epilogue, write [b,h,t,d] bf16 ----
    #pragma unroll
    for (int nf = 0; nf < 4; nf++) {
        #pragma unroll
        for (int r = 0; r < 8; r++) {
            int row = m0 + w * 16 + r + half * 8;
            int col = n0 + nf * 16 + l16;
            int b = row >> 11, t = row & (TT - 1);
            int h = col >> 6,  d = col & (DD - 1);
            float c = cosp[t * DD + d], s = sinp[t * DD + d];
            float re = accRe[nf][r], im = accIm[nf][r];
            int idx = ((b * HH + h) * TT + t) * DD + d;
            outRe[idx] = (__bf16)((re * c - im * s) * outScale);
            outIm[idx] = (__bf16)((re * s + im * c) * outScale);
        }
    }
}

// ---------------------------------------------------------------------------
// V projection: outVt[b,h,d,t] = (x @ Wv)^T per head  (bf16, d-major so the
// attention kernel can async-stage V tiles). Packed 16B epilogue stores.
// ---------------------------------------------------------------------------
__global__ __launch_bounds__(128)
void proj_v_kernel(const __bf16* __restrict__ xb,
                   const __bf16* __restrict__ WvT,
                   __bf16* __restrict__ outVt)
{
    __shared__ __align__(16) __bf16 AsB[2][64 * 32];
    __shared__ __align__(16) __bf16 BtB[2][64 * 32];

    const int tid  = threadIdx.x;
    const int lane = tid & 31, w = tid >> 5;
    const int half = lane >> 4, l16 = lane & 15;
    const int m0 = blockIdx.x * 64;
    const int n0 = blockIdx.y * 64;

    const int c0 = tid, c1 = tid + 128;
    const int ar0 = c0 >> 2, ak0 = (c0 & 3) << 3;
    const int ar1 = c1 >> 2, ak1 = (c1 & 3) << 3;

    v8f acc[4];
    #pragma unroll
    for (int i = 0; i < 4; i++) acc[i] = vzero8();

    async_cp16(&xb [(m0 + ar0) * DM + ak0], &AsB[0][ar0 * 32 + ak0]);
    async_cp16(&xb [(m0 + ar1) * DM + ak1], &AsB[0][ar1 * 32 + ak1]);
    async_cp16(&WvT[(n0 + ar0) * DM + ak0], &BtB[0][ar0 * 32 + ak0]);
    async_cp16(&WvT[(n0 + ar1) * DM + ak1], &BtB[0][ar1 * 32 + ak1]);

    for (int kb = 0; kb < DM; kb += 32) {
        const int buf = (kb >> 5) & 1;
        wait_async0();
        __syncthreads();
        if (kb + 32 < DM) {
            const int nb = buf ^ 1, kn = kb + 32;
            async_cp16(&xb [(m0 + ar0) * DM + kn + ak0], &AsB[nb][ar0 * 32 + ak0]);
            async_cp16(&xb [(m0 + ar1) * DM + kn + ak1], &AsB[nb][ar1 * 32 + ak1]);
            async_cp16(&WvT[(n0 + ar0) * DM + kn + ak0], &BtB[nb][ar0 * 32 + ak0]);
            async_cp16(&WvT[(n0 + ar1) * DM + kn + ak1], &BtB[nb][ar1 * 32 + ak1]);
        }
        const __bf16* A = AsB[buf];
        const __bf16* Bt = BtB[buf];

        FragA a;
        a.q[0] = lds16(&A[(w * 16 + l16) * 32 + half * 8]);
        a.q[1] = lds16(&A[(w * 16 + l16) * 32 + half * 8 + 16]);
        FragA fb[4];
        #pragma unroll
        for (int nf = 0; nf < 4; nf++) {
            const int nn = nf * 16 + l16;
            fb[nf].q[0] = lds16(&Bt[nn * 32 + half * 8]);
            fb[nf].q[1] = lds16(&Bt[nn * 32 + half * 8 + 16]);
        }
        #pragma unroll
        for (int nf = 0; nf < 4; nf++) acc[nf] = wmma_bf16(a, fb[nf], acc[nf]);
    }

    // ---- packed transposed store: 8 consecutive t per lane -> one 16B store
    const int b  = m0 >> 11;
    const int tb = (m0 & (TT - 1)) + w * 16 + half * 8;
    #pragma unroll
    for (int nf = 0; nf < 4; nf++) {
        int col = n0 + nf * 16 + l16;
        int h = col >> 6, d = col & (DD - 1);
        U8 u;
        #pragma unroll
        for (int r = 0; r < 8; r++) u.b[r] = (__bf16)acc[nf][r];
        *reinterpret_cast<uint4*>(&outVt[((size_t)(b * HH + h) * DD + d) * TT + tb]) = u.u;
    }
}

// ---------------------------------------------------------------------------
// Causal flash attention (Hermitian scores = qre.kre + qim.kim, scale folded
// into q). 128 thr / block, 64 q-rows, 32 keys per tile, async double-buffer.
// Output bf16 in [b, t, h*d] for the final projection.
// ---------------------------------------------------------------------------
__global__ __launch_bounds__(128)
void attn_kernel(const __bf16* __restrict__ qre, const __bf16* __restrict__ qim,
                 const __bf16* __restrict__ kre, const __bf16* __restrict__ kim,
                 const __bf16* __restrict__ vt,  __bf16* __restrict__ attnb)
{
    __shared__ __align__(16) __bf16 KreB[2][32 * 64];   // [key][d]
    __shared__ __align__(16) __bf16 KimB[2][32 * 64];
    __shared__ __align__(16) __bf16 VtB [2][64 * 32];   // [d][key]
    __shared__ __align__(16) __bf16 Plds[4 * 16 * 32];  // per-wave [row][key]

    const int tid  = threadIdx.x;
    const int lane = tid & 31, w = tid >> 5;
    const int half = lane >> 4, l16 = lane & 15;
    const int qt = blockIdx.x;
    const int bh = blockIdx.y;
    const int qbase = qt * 64;

    const __bf16* qreP = qre + (size_t)bh * TT * DD;
    const __bf16* qimP = qim + (size_t)bh * TT * DD;
    const __bf16* kreP = kre + (size_t)bh * TT * DD;
    const __bf16* kimP = kim + (size_t)bh * TT * DD;
    const __bf16* vtP  = vt  + (size_t)bh * TT * DD;   // [d][t]

    // chunk ids for K staging ([key][d]): key = c>>3, doff = (c&7)*8
    const int kc0 = tid, kc1 = tid + 128;
    const int kk0 = kc0 >> 3, kd0 = (kc0 & 7) << 3;
    const int kk1 = kc1 >> 3, kd1 = (kc1 & 7) << 3;
    // chunk ids for V staging ([d][key]): d = c>>2, koff = (c&3)*8
    const int vd0 = kc0 >> 2, vk0 = (kc0 & 3) << 3;
    const int vd1 = kc1 >> 2, vk1 = (kc1 & 3) << 3;

    // Q fragments (A-layout), 2 chunks of 32 over D=64
    FragA aRe[2], aIm[2];
    {
        int trow = qbase + w * 16 + l16;
        #pragma unroll
        for (int c = 0; c < 2; c++) {
            int d0 = c * 32 + half * 8;
            aRe[c].q[0] = ldg16(&qreP[trow * DD + d0]);
            aRe[c].q[1] = ldg16(&qreP[trow * DD + d0 + 16]);
            aIm[c].q[0] = ldg16(&qimP[trow * DD + d0]);
            aIm[c].q[1] = ldg16(&qimP[trow * DD + d0 + 16]);
        }
    }

    v8f accO[4];
    #pragma unroll
    for (int i = 0; i < 4; i++) accO[i] = vzero8();
    float mrow[8], lrow[8];
    #pragma unroll
    for (int r = 0; r < 8; r++) { mrow[r] = -3.0e38f; lrow[r] = 0.0f; }

    const int nkt = (qbase + 64) / 32;

    // stage tile 0
    {
        async_cp16(&kreP[(0 + kk0) * DD + kd0], &KreB[0][kk0 * 64 + kd0]);
        async_cp16(&kreP[(0 + kk1) * DD + kd1], &KreB[0][kk1 * 64 + kd1]);
        async_cp16(&kimP[(0 + kk0) * DD + kd0], &KimB[0][kk0 * 64 + kd0]);
        async_cp16(&kimP[(0 + kk1) * DD + kd1], &KimB[0][kk1 * 64 + kd1]);
        async_cp16(&vtP[vd0 * TT + 0 + vk0],    &VtB [0][vd0 * 32 + vk0]);
        async_cp16(&vtP[vd1 * TT + 0 + vk1],    &VtB [0][vd1 * 32 + vk1]);
    }

    for (int kt = 0; kt < nkt; kt++) {
        const int ktb = kt * 32;
        const int buf = kt & 1;
        wait_async0();
        __syncthreads();
        if (kt + 1 < nkt) {
            const int nb = buf ^ 1, kn = ktb + 32;
            async_cp16(&kreP[(kn + kk0) * DD + kd0], &KreB[nb][kk0 * 64 + kd0]);
            async_cp16(&kreP[(kn + kk1) * DD + kd1], &KreB[nb][kk1 * 64 + kd1]);
            async_cp16(&kimP[(kn + kk0) * DD + kd0], &KimB[nb][kk0 * 64 + kd0]);
            async_cp16(&kimP[(kn + kk1) * DD + kd1], &KimB[nb][kk1 * 64 + kd1]);
            async_cp16(&vtP[vd0 * TT + kn + vk0],    &VtB [nb][vd0 * 32 + vk0]);
            async_cp16(&vtP[vd1 * TT + kn + vk1],    &VtB [nb][vd1 * 32 + vk1]);
        }
        const __bf16* Kre = KreB[buf];
        const __bf16* Kim = KimB[buf];
        const __bf16* Vt  = VtB [buf];

        // ---- scores: S = qre.kre^T + qim.kim^T ----
        FragA kR[2][2], kI[2][2];
        #pragma unroll
        for (int nf = 0; nf < 2; nf++) {
            int key = nf * 16 + l16;
            #pragma unroll
            for (int c = 0; c < 2; c++) {
                int d0 = c * 32 + half * 8;
                kR[nf][c].q[0] = lds16(&Kre[key * 64 + d0]);
                kR[nf][c].q[1] = lds16(&Kre[key * 64 + d0 + 16]);
                kI[nf][c].q[0] = lds16(&Kim[key * 64 + d0]);
                kI[nf][c].q[1] = lds16(&Kim[key * 64 + d0 + 16]);
            }
        }
        v8f accS[2]; accS[0] = vzero8(); accS[1] = vzero8();
        #pragma unroll
        for (int nf = 0; nf < 2; nf++) {
            #pragma unroll
            for (int c = 0; c < 2; c++) {
                accS[nf] = wmma_bf16(aRe[c], kR[nf][c], accS[nf]);
                accS[nf] = wmma_bf16(aIm[c], kI[nf][c], accS[nf]);
            }
        }

        // ---- causal mask + online softmax ----
        float p[2][8], mnew[8];
        #pragma unroll
        for (int r = 0; r < 8; r++) {
            int qrow = qbase + w * 16 + r + half * 8;
            float s0 = (ktb +      l16 <= qrow) ? accS[0][r] : -3.0e38f;
            float s1 = (ktb + 16 + l16 <= qrow) ? accS[1][r] : -3.0e38f;
            p[0][r] = s0; p[1][r] = s1;
            float mr = fmaxf(s0, s1);
            #pragma unroll
            for (int off = 8; off; off >>= 1) mr = fmaxf(mr, __shfl_xor(mr, off));
            mnew[r] = fmaxf(mrow[r], mr);
        }
        #pragma unroll
        for (int r = 0; r < 8; r++) {
            float alpha = exp2f((mrow[r] - mnew[r]) * LOG2E);
            float p0 = exp2f((p[0][r] - mnew[r]) * LOG2E);
            float p1 = exp2f((p[1][r] - mnew[r]) * LOG2E);
            p[0][r] = p0; p[1][r] = p1;
            float ps = p0 + p1;
            #pragma unroll
            for (int off = 8; off; off >>= 1) ps += __shfl_xor(ps, off);
            lrow[r] = lrow[r] * alpha + ps;
            mrow[r] = mnew[r];
            #pragma unroll
            for (int j = 0; j < 4; j++) accO[j][r] *= alpha;
        }

        // ---- P: C-layout -> A-layout via per-wave LDS round trip ----
        __bf16* P = &Plds[w * 512];
        #pragma unroll
        for (int nf = 0; nf < 2; nf++)
            #pragma unroll
            for (int r = 0; r < 8; r++)
                P[(r + half * 8) * 32 + nf * 16 + l16] = (__bf16)p[nf][r];

        FragA pf;
        pf.q[0] = lds16(&P[l16 * 32 + half * 8]);
        pf.q[1] = lds16(&P[l16 * 32 + half * 8 + 16]);

        // ---- O += P @ V ----
        FragA bv[4];
        #pragma unroll
        for (int j = 0; j < 4; j++) {
            int dd = j * 16 + l16;
            bv[j].q[0] = lds16(&Vt[dd * 32 + half * 8]);
            bv[j].q[1] = lds16(&Vt[dd * 32 + half * 8 + 16]);
        }
        #pragma unroll
        for (int j = 0; j < 4; j++) accO[j] = wmma_bf16(pf, bv[j], accO[j]);
    }

    // ---- normalize and store [b, t, h*D] bf16 ----
    const int b = bh / HH, h = bh % HH;
    float invl[8];
    #pragma unroll
    for (int r = 0; r < 8; r++) invl[r] = 1.0f / lrow[r];
    #pragma unroll
    for (int j = 0; j < 4; j++) {
        #pragma unroll
        for (int r = 0; r < 8; r++) {
            int row = b * TT + qbase + w * 16 + r + half * 8;
            attnb[(size_t)row * HD + h * DD + j * 16 + l16] =
                (__bf16)(accO[j][r] * invl[r]);
        }
    }
}

// ---------------------------------------------------------------------------
// Output projection: out[f32] = attn[bf16, 4096x1024] @ Wo (pre-transposed)
// ---------------------------------------------------------------------------
__global__ __launch_bounds__(128)
void out_proj_kernel(const __bf16* __restrict__ attnb,
                     const __bf16* __restrict__ WoT,
                     float* __restrict__ out)
{
    __shared__ __align__(16) __bf16 AsB[2][64 * 32];
    __shared__ __align__(16) __bf16 BtB[2][64 * 32];

    const int tid  = threadIdx.x;
    const int lane = tid & 31, w = tid >> 5;
    const int half = lane >> 4, l16 = lane & 15;
    const int m0 = blockIdx.x * 64;
    const int n0 = blockIdx.y * 64;

    const int c0 = tid, c1 = tid + 128;
    const int ar0 = c0 >> 2, ak0 = (c0 & 3) << 3;
    const int ar1 = c1 >> 2, ak1 = (c1 & 3) << 3;

    v8f acc[4];
    #pragma unroll
    for (int i = 0; i < 4; i++) acc[i] = vzero8();

    async_cp16(&attnb[(m0 + ar0) * HD + ak0], &AsB[0][ar0 * 32 + ak0]);
    async_cp16(&attnb[(m0 + ar1) * HD + ak1], &AsB[0][ar1 * 32 + ak1]);
    async_cp16(&WoT[(n0 + ar0) * HD + ak0],   &BtB[0][ar0 * 32 + ak0]);
    async_cp16(&WoT[(n0 + ar1) * HD + ak1],   &BtB[0][ar1 * 32 + ak1]);

    for (int kb = 0; kb < HD; kb += 32) {
        const int buf = (kb >> 5) & 1;
        wait_async0();
        __syncthreads();
        if (kb + 32 < HD) {
            const int nb = buf ^ 1, kn = kb + 32;
            async_cp16(&attnb[(m0 + ar0) * HD + kn + ak0], &AsB[nb][ar0 * 32 + ak0]);
            async_cp16(&attnb[(m0 + ar1) * HD + kn + ak1], &AsB[nb][ar1 * 32 + ak1]);
            async_cp16(&WoT[(n0 + ar0) * HD + kn + ak0],   &BtB[nb][ar0 * 32 + ak0]);
            async_cp16(&WoT[(n0 + ar1) * HD + kn + ak1],   &BtB[nb][ar1 * 32 + ak1]);
        }
        const __bf16* A  = AsB[buf];
        const __bf16* Bt = BtB[buf];

        FragA a;
        a.q[0] = lds16(&A[(w * 16 + l16) * 32 + half * 8]);
        a.q[1] = lds16(&A[(w * 16 + l16) * 32 + half * 8 + 16]);
        FragA fb[4];
        #pragma unroll
        for (int nf = 0; nf < 4; nf++) {
            const int nn = nf * 16 + l16;
            fb[nf].q[0] = lds16(&Bt[nn * 32 + half * 8]);
            fb[nf].q[1] = lds16(&Bt[nn * 32 + half * 8 + 16]);
        }
        #pragma unroll
        for (int nf = 0; nf < 4; nf++) acc[nf] = wmma_bf16(a, fb[nf], acc[nf]);
    }

    #pragma unroll
    for (int nf = 0; nf < 4; nf++)
        #pragma unroll
        for (int r = 0; r < 8; r++) {
            int row = m0 + w * 16 + r + half * 8;
            int col = n0 + nf * 16 + l16;
            out[(size_t)row * DM + col] = acc[nf][r];
        }
}

// ---------------------------------------------------------------------------
// Host launcher
// ---------------------------------------------------------------------------
extern "C" void kernel_launch(void* const* d_in, const int* in_sizes, int n_in,
                              void* d_out, int out_size, void* d_ws, size_t ws_size,
                              hipStream_t stream) {
    (void)in_sizes; (void)n_in; (void)out_size; (void)ws_size;

    const float* x     = (const float*)d_in[0];
    const float* Wq_re = (const float*)d_in[1];
    const float* Wq_im = (const float*)d_in[2];
    const float* Wk_re = (const float*)d_in[3];
    const float* Wk_im = (const float*)d_in[4];
    const float* Wv    = (const float*)d_in[5];
    const float* Wo    = (const float*)d_in[6];
    const float* cosp  = (const float*)d_in[7];
    const float* sinp  = (const float*)d_in[8];
    float* out = (float*)d_out;

    const size_t NX = (size_t)MROWS * DM;        // 4,194,304
    const size_t NW = (size_t)DM * HD;           // 1,048,576
    const size_t NH = (size_t)BB * HH * TT * DD; // 4,194,304

    char* p = (char*)d_ws;
    __bf16* xb    = (__bf16*)p; p += NX * 2;
    __bf16* wqreT = (__bf16*)p; p += NW * 2;
    __bf16* wqimT = (__bf16*)p; p += NW * 2;
    __bf16* wkreT = (__bf16*)p; p += NW * 2;
    __bf16* wkimT = (__bf16*)p; p += NW * 2;
    __bf16* wvT   = (__bf16*)p; p += NW * 2;
    __bf16* woT   = (__bf16*)p; p += NW * 2;
    __bf16* qreR  = (__bf16*)p; p += NH * 2;
    __bf16* qimR  = (__bf16*)p; p += NH * 2;
    __bf16* kreR  = (__bf16*)p; p += NH * 2;
    __bf16* kimR  = (__bf16*)p; p += NH * 2;
    __bf16* vT    = (__bf16*)p; p += NH * 2;
    __bf16* attnb = (__bf16*)p; p += NX * 2;

    // 1) conversions (+ weight transpose so GEMM staging is pure async copies)
    cvt_bf16_kernel<<<(int)((NX + 255) / 256), 256, 0, stream>>>(x, xb, (int)NX);
    dim3 gt(32, 32);
    cvt_t_kernel<<<gt, 256, 0, stream>>>(Wq_re, wqreT);
    cvt_t_kernel<<<gt, 256, 0, stream>>>(Wq_im, wqimT);
    cvt_t_kernel<<<gt, 256, 0, stream>>>(Wk_re, wkreT);
    cvt_t_kernel<<<gt, 256, 0, stream>>>(Wk_im, wkimT);
    cvt_t_kernel<<<gt, 256, 0, stream>>>(Wv, wvT);
    cvt_t_kernel<<<gt, 256, 0, stream>>>(Wo, woT);

    // 2) projections + RoPE (scale folded into Q)
    dim3 gp(MROWS / 64, HD / 64);
    proj_pair_rope_kernel<<<gp, 128, 0, stream>>>(xb, wqreT, wqimT, cosp, sinp,
                                                  qreR, qimR, SCALE);
    proj_pair_rope_kernel<<<gp, 128, 0, stream>>>(xb, wkreT, wkimT, cosp, sinp,
                                                  kreR, kimR, 1.0f);
    proj_v_kernel<<<gp, 128, 0, stream>>>(xb, wvT, vT);

    // 3) causal flash attention
    dim3 ga(TT / 64, BB * HH);
    attn_kernel<<<ga, 128, 0, stream>>>(qreR, qimR, kreR, kimR, vT, attnb);

    // 4) output projection -> f32
    dim3 go(MROWS / 64, DM / 64);
    out_proj_kernel<<<go, 128, 0, stream>>>(attnb, woT, out);
}
